// TransBlock_29729763623677
// MI455X (gfx1250) — compile-verified, matrix operand order checked by reading, not run on hardware
//
#include <hip/hip_runtime.h>
#include <math.h>

// ---------------------------------------------------------------------------
// Problem constants (B=1)
// ---------------------------------------------------------------------------
#define H_     192
#define W_     192
#define P_     36864          // H_*W_
#define C_     48             // DIM
#define HD_    24             // head dim
#define NQKV_  144            // 3*DIM
#define HID_   127            // int(48*2.66)
#define CH2_   254            // 2*HID
#define KWIN   7
#define DIL_   3

// padded strides for WMMA-fed f16 buffers (16B-aligned rows, zero-filled tails)
#define APAD   64             // channel stride for 48-ch f16 activations
#define GPAD   128            // channel stride for 127-ch gated buffer
#define TPAD   256            // channel stride for 254-ch FFN hidden buffer

typedef __attribute__((ext_vector_type(16))) _Float16 v16h;
typedef __attribute__((ext_vector_type(8)))  _Float16 v8h;
typedef __attribute__((ext_vector_type(8)))  float    v8f;

// ---------------------------------------------------------------------------
// Branch-free WMMA 16x16x32 f16 fragment load (CDNA5 layout, ISA 7.12.2).
// For both A (row = M) and B (row = N): lane%16 selects the row, lanes>=16
// take k+8; element e covers k = kb + e (e<8) and k = kb + 16 + (e-8).
// Rows are 16B aligned (ld is a multiple of 8 halves), so this is exactly
// two global_load_b128.
// ---------------------------------------------------------------------------
__device__ __forceinline__ v16h load_frag(const _Float16* __restrict__ p, int ld,
                                          int row, int k0, int lane) {
  const int kb = (lane & 16) ? 8 : 0;
  const _Float16* base = p + row * ld + k0 + kb;
  v8h lo = *(const v8h*)(base);
  v8h hi = *(const v8h*)(base + 16);
  return __builtin_shufflevector(lo, hi, 0, 1, 2, 3, 4, 5, 6, 7,
                                 8, 9, 10, 11, 12, 13, 14, 15);
}

// 64x16 output tile per wave: 4 stacked M-tiles sharing one B fragment.
template <int KPAD>
__device__ __forceinline__ void gemm64x16(const _Float16* __restrict__ A, int lda,
                                          const _Float16* __restrict__ B, int ldb,
                                          int m0, int n0, int lane, v8f acc[4]) {
#pragma unroll
  for (int k0 = 0; k0 < KPAD; k0 += 32) {
    v16h b = load_frag(B, ldb, n0 + (lane & 15), k0, lane);
#pragma unroll
    for (int t = 0; t < 4; ++t) {
      v16h a = load_frag(A, lda, m0 + t * 16 + (lane & 15), k0, lane);
      acc[t] = __builtin_amdgcn_wmma_f32_16x16x32_f16(false, a, false, b, (short)0,
                                                      acc[t], false, false);
    }
  }
}

// ---------------------------------------------------------------------------
// 0. Weight conversion: f32 (N,K) row-major -> f16 padded (Npad,Kpad), zero fill
// ---------------------------------------------------------------------------
__global__ void k_convw(const float* __restrict__ W, _Float16* __restrict__ out,
                        int N, int K, int Kpad, int total) {
  const int idx = blockIdx.x * blockDim.x + threadIdx.x;
  if (idx >= total) return;
  const int n = idx / Kpad, k = idx - n * Kpad;
  out[idx] = (n < N && k < K) ? (_Float16)W[n * K + k] : (_Float16)0.0f;
}

// ---------------------------------------------------------------------------
// 1. LayerNorm over channels: x NCHW f32 -> xn1 NHWC f16 (stride APAD, padded)
// ---------------------------------------------------------------------------
__global__ void k_ln1(const float* __restrict__ x, const float* __restrict__ w,
                      const float* __restrict__ b, _Float16* __restrict__ xn) {
  const int p = blockIdx.x * blockDim.x + threadIdx.x;
  if (p >= P_) return;
  float xv[C_];
  float s = 0.f, ss = 0.f;
#pragma unroll
  for (int c = 0; c < C_; ++c) { float v = x[c * P_ + p]; xv[c] = v; s += v; ss += v * v; }
  const float mu  = s * (1.0f / C_);
  const float var = ss * (1.0f / C_) - mu * mu;
  const float r   = rsqrtf(var + 1e-5f);
#pragma unroll
  for (int c = 0; c < C_; ++c)
    xn[p * APAD + c] = (_Float16)((xv[c] - mu) * r * w[c] + b[c]);
#pragma unroll
  for (int c = C_; c < APAD; ++c) xn[p * APAD + c] = (_Float16)0.0f;
}

// LN2: x_mid NHWC f32 -> xn2 NHWC f16 (stride APAD, padded)
__global__ void k_ln2(const float* __restrict__ xm, const float* __restrict__ w,
                      const float* __restrict__ b, _Float16* __restrict__ xn) {
  const int p = blockIdx.x * blockDim.x + threadIdx.x;
  if (p >= P_) return;
  float xv[C_];
  float s = 0.f, ss = 0.f;
#pragma unroll
  for (int c = 0; c < C_; ++c) { float v = xm[p * C_ + c]; xv[c] = v; s += v; ss += v * v; }
  const float mu  = s * (1.0f / C_);
  const float var = ss * (1.0f / C_) - mu * mu;
  const float r   = rsqrtf(var + 1e-5f);
#pragma unroll
  for (int c = 0; c < C_; ++c)
    xn[p * APAD + c] = (_Float16)((xv[c] - mu) * r * w[c] + b[c]);
#pragma unroll
  for (int c = C_; c < APAD; ++c) xn[p * APAD + c] = (_Float16)0.0f;
}

// ---------------------------------------------------------------------------
// 2. QKV GEMM: [P,48] f16 x [48,144] -> qkv [P,144] f32 (+bias)
// ---------------------------------------------------------------------------
__global__ void k_gemm_qkv(const _Float16* __restrict__ A, const _Float16* __restrict__ Bh,
                           const float* __restrict__ bias, float* __restrict__ out) {
  const int wave = blockIdx.x * 8 + (threadIdx.x >> 5);
  const int lane = threadIdx.x & 31;
  const int NT = NQKV_ / 16;                    // 9
  const int nt = wave % NT;
  const int m0 = (wave / NT) * 64;
  v8f acc[4] = {};
  gemm64x16<64>(A, APAD, Bh, 64, m0, nt * 16, lane, acc);
  const int nl = lane & 15, mb = (lane & 16) ? 8 : 0;
  const int n  = nt * 16 + nl;
  const float bi = bias[n];
#pragma unroll
  for (int t = 0; t < 4; ++t)
#pragma unroll
    for (int i = 0; i < 8; ++i)
      out[(m0 + t * 16 + mb + i) * NQKV_ + n] = acc[t][i] + bi;
}

// ---------------------------------------------------------------------------
// 3. Neighborhood attention (K=7, dil=3). Thread = (pixel, head).
// ---------------------------------------------------------------------------
__device__ __forceinline__ int win_start(int i) {     // L=192, K=7, D=3
  int st = i - 9;                                     // i - (K//2)*D
  if (st < 0) st = i % 3;
  else if (i + 9 >= 192) st = 171 + (i % 3);          // L - K*D + i%D (b==0 case)
  return st;
}

__global__ void k_attn(const float* __restrict__ qkv, const float* __restrict__ rpb,
                       _Float16* __restrict__ out) {
  __shared__ float s_a[256 * 49];                     // 50176 B
  const int t = blockIdx.x * blockDim.x + threadIdx.x;  // exact: P_*2 threads
  const int p = t >> 1;
  const int head = t & 1;
  const int i = p / W_, j = p - i * W_;
  const int sti = win_start(i), stj = win_start(j);
  const int pbi = (sti - i) / DIL_ + (KWIN - 1);      // exact division (multiple of 3)
  const int pbj = (stj - j) / DIL_ + (KWIN - 1);
  const float* qp = qkv + p * NQKV_ + head * HD_;
  float q[HD_];
#pragma unroll
  for (int c = 0; c < HD_; ++c) q[c] = qp[c] * 0.20412414523193154f;  // HD^-0.5
  float* aw = s_a + threadIdx.x * 49;
  const float* rb = rpb + head * (13 * 13);
  float mx = -1e30f;
  for (int x = 0; x < KWIN; ++x) {
    const int ih = sti + x * DIL_;
    for (int y = 0; y < KWIN; ++y) {
      const int iw = stj + y * DIL_;
      const float* kp = qkv + (ih * W_ + iw) * NQKV_ + C_ + head * HD_;
      float d = 0.f;
#pragma unroll
      for (int c = 0; c < HD_; ++c) d += q[c] * kp[c];
      d += rb[(pbi + x) * 13 + (pbj + y)];
      aw[x * KWIN + y] = d;
      mx = fmaxf(mx, d);
    }
  }
  float s = 0.f;
  for (int e = 0; e < 49; ++e) { float v = expf(aw[e] - mx); aw[e] = v; s += v; }
  const float inv = 1.0f / s;
  float acc[HD_];
#pragma unroll
  for (int c = 0; c < HD_; ++c) acc[c] = 0.f;
  for (int x = 0; x < KWIN; ++x) {
    const int ih = sti + x * DIL_;
    for (int y = 0; y < KWIN; ++y) {
      const int iw = stj + y * DIL_;
      const float wgt = aw[x * KWIN + y] * inv;
      const float* vp = qkv + (ih * W_ + iw) * NQKV_ + 2 * C_ + head * HD_;
#pragma unroll
      for (int c = 0; c < HD_; ++c) acc[c] += wgt * vp[c];
    }
  }
  _Float16* op = out + p * APAD + head * HD_;
#pragma unroll
  for (int c = 0; c < HD_; ++c) op[c] = (_Float16)acc[c];
  if (head) {                       // zero the padded tail of this pixel's row
#pragma unroll
    for (int c = C_; c < APAD; ++c) out[p * APAD + c] = (_Float16)0.0f;
  }
}

// ---------------------------------------------------------------------------
// 4/5. ECA: global mean pool per channel, then 7-tap conv1d + sigmoid -> gate
// ---------------------------------------------------------------------------
__global__ void k_eca_pool(const _Float16* __restrict__ xn1, float* __restrict__ pooled) {
  __shared__ float red[256];
  const int c = blockIdx.x;
  float s = 0.f;
  for (int p = threadIdx.x; p < P_; p += 256) s += (float)xn1[p * APAD + c];
  red[threadIdx.x] = s;
  __syncthreads();
  for (int st = 128; st > 0; st >>= 1) {
    if ((int)threadIdx.x < st) red[threadIdx.x] += red[threadIdx.x + st];
    __syncthreads();
  }
  if (threadIdx.x == 0) pooled[c] = red[0] * (1.0f / P_);
}

__global__ void k_eca_gate(const float* __restrict__ pooled, const float* __restrict__ w,
                           float* __restrict__ gate) {
  const int c = threadIdx.x;
  if (c >= C_) return;
  float y = 0.f;
#pragma unroll
  for (int t = 0; t < 7; ++t) {
    const int cc = c + t - 3;
    if (cc >= 0 && cc < C_) y += pooled[cc] * w[t];
  }
  gate[c] = 1.0f / (1.0f + expf(-y));
}

// ---------------------------------------------------------------------------
// 6. Proj GEMM + bias + channel gate + residual: x_mid[p][c] (NHWC f32)
// ---------------------------------------------------------------------------
__global__ void k_gemm_proj(const _Float16* __restrict__ A, const _Float16* __restrict__ Bh,
                            const float* __restrict__ bias, const float* __restrict__ gate,
                            const float* __restrict__ x_nchw, float* __restrict__ xmid) {
  const int wave = blockIdx.x * 8 + (threadIdx.x >> 5);
  const int lane = threadIdx.x & 31;
  const int NT = C_ / 16;                        // 3
  const int nt = wave % NT;
  const int m0 = (wave / NT) * 64;
  v8f acc[4] = {};
  gemm64x16<64>(A, APAD, Bh, 64, m0, nt * 16, lane, acc);
  const int nl = lane & 15, mb = (lane & 16) ? 8 : 0;
  const int n  = nt * 16 + nl;
  const float g = gate[n], bi = bias[n];
#pragma unroll
  for (int t = 0; t < 4; ++t)
#pragma unroll
    for (int i = 0; i < 8; ++i) {
      const int m = m0 + t * 16 + mb + i;
      xmid[m * C_ + n] = x_nchw[n * P_ + m] + (acc[t][i] + bi) * g;
    }
}

// ---------------------------------------------------------------------------
// 8. FFN up-proj GEMM: [P,48] f16 x [48,256pad] -> t [P,TPAD] f16
//    (B rows 254/255 are zero, so pad columns of t get exact zeros)
// ---------------------------------------------------------------------------
__global__ void k_gemm_win(const _Float16* __restrict__ A, const _Float16* __restrict__ Bh,
                           _Float16* __restrict__ out) {
  const int wave = blockIdx.x * 8 + (threadIdx.x >> 5);
  const int lane = threadIdx.x & 31;
  const int NT = TPAD / 16;                      // 16
  const int nt = wave % NT;
  const int m0 = (wave / NT) * 64;
  v8f acc[4] = {};
  gemm64x16<64>(A, APAD, Bh, 64, m0, nt * 16, lane, acc);
  const int nl = lane & 15, mb = (lane & 16) ? 8 : 0;
  const int n  = nt * 16 + nl;
#pragma unroll
  for (int t = 0; t < 4; ++t)
#pragma unroll
    for (int i = 0; i < 8; ++i)
      out[(m0 + t * 16 + mb + i) * TPAD + n] = (_Float16)acc[t][i];
}

// ---------------------------------------------------------------------------
// 9. Depthwise 3x3 (pad 1) + exact-GELU gate: g[p][ch] f16, stride GPAD
// ---------------------------------------------------------------------------
__global__ void k_dwgelu(const _Float16* __restrict__ t, const float* __restrict__ wdw,
                         _Float16* __restrict__ g) {
  const int idx = blockIdx.x * blockDim.x + threadIdx.x;   // exact grid: P_*HID_
  const int p = idx / HID_, ch = idx - p * HID_;
  const int i = p / W_, j = p - i * W_;
  float a1 = 0.f, a2 = 0.f;
  const float* w1 = wdw + ch * 9;
  const float* w2 = wdw + (ch + HID_) * 9;
#pragma unroll
  for (int di = 0; di < 3; ++di) {
    const int ii = i + di - 1;
    if ((unsigned)ii >= (unsigned)H_) continue;
#pragma unroll
    for (int dj = 0; dj < 3; ++dj) {
      const int jj = j + dj - 1;
      if ((unsigned)jj >= (unsigned)W_) continue;
      const _Float16* tp = t + (ii * W_ + jj) * TPAD;
      a1 += (float)tp[ch]        * w1[di * 3 + dj];
      a2 += (float)tp[ch + HID_] * w2[di * 3 + dj];
    }
  }
  const float gl = 0.5f * a1 * (1.0f + erff(a1 * 0.70710678118654752f));
  g[p * GPAD + ch] = (_Float16)(gl * a2);
  if (ch == 0) g[p * GPAD + HID_] = (_Float16)0.0f;   // zero the pad column
}

// ---------------------------------------------------------------------------
// 10. FFN down-proj GEMM (Kpad=128) + residual, store NCHW f32
// ---------------------------------------------------------------------------
__global__ void k_gemm_wout(const _Float16* __restrict__ A, const _Float16* __restrict__ Bh,
                            const float* __restrict__ xmid, float* __restrict__ out) {
  const int wave = blockIdx.x * 8 + (threadIdx.x >> 5);
  const int lane = threadIdx.x & 31;
  const int NT = C_ / 16;                        // 3
  const int nt = wave % NT;
  const int m0 = (wave / NT) * 64;
  v8f acc[4] = {};
  gemm64x16<128>(A, GPAD, Bh, 128, m0, nt * 16, lane, acc);
  const int nl = lane & 15, mb = (lane & 16) ? 8 : 0;
  const int n  = nt * 16 + nl;
#pragma unroll
  for (int t = 0; t < 4; ++t)
#pragma unroll
    for (int i = 0; i < 8; ++i) {
      const int m = m0 + t * 16 + mb + i;
      out[n * P_ + m] = xmid[m * C_ + n] + acc[t][i];
    }
}

// ---------------------------------------------------------------------------
// Host launcher
// ---------------------------------------------------------------------------
extern "C" void kernel_launch(void* const* d_in, const int* in_sizes, int n_in,
                              void* d_out, int out_size, void* d_ws, size_t ws_size,
                              hipStream_t stream) {
  const float* x      = (const float*)d_in[0];
  const float* ln1_w  = (const float*)d_in[1];
  const float* ln1_b  = (const float*)d_in[2];
  const float* qkv_w  = (const float*)d_in[3];
  const float* qkv_b  = (const float*)d_in[4];
  const float* rpb    = (const float*)d_in[5];
  const float* proj_w = (const float*)d_in[6];
  const float* proj_b = (const float*)d_in[7];
  const float* eca_w  = (const float*)d_in[8];
  const float* ln2_w  = (const float*)d_in[9];
  const float* ln2_b  = (const float*)d_in[10];
  const float* w_in   = (const float*)d_in[11];
  const float* w_dw   = (const float*)d_in[12];
  const float* w_out  = (const float*)d_in[13];
  float* out = (float*)d_out;

  // workspace carve-out (256B aligned)
  char* ws = (char*)d_ws;
  size_t o = 0;
  auto carve = [&](size_t bytes) -> char* {
    char* r = ws + o;
    o = (o + bytes + 255) & ~(size_t)255;
    return r;
  };
  _Float16* xn1   = (_Float16*)carve((size_t)P_ * APAD * 2);
  float*    qkv   = (float*)   carve((size_t)P_ * NQKV_ * 4);
  _Float16* attn  = (_Float16*)carve((size_t)P_ * APAD * 2);
  float*    xmid  = (float*)   carve((size_t)P_ * C_ * 4);
  _Float16* xn2   = (_Float16*)carve((size_t)P_ * APAD * 2);
  _Float16* t_h   = (_Float16*)carve((size_t)P_ * TPAD * 2);
  _Float16* g_h   = (_Float16*)carve((size_t)P_ * GPAD * 2);
  _Float16* qkvwh = (_Float16*)carve((size_t)NQKV_ * 64 * 2);
  _Float16* projwh= (_Float16*)carve((size_t)C_ * 64 * 2);
  _Float16* winwh = (_Float16*)carve((size_t)TPAD * 64 * 2);
  _Float16* woutwh= (_Float16*)carve((size_t)C_ * 128 * 2);
  float*    pooled= (float*)   carve(C_ * 4);
  float*    gate  = (float*)   carve(C_ * 4);
  (void)ws_size; (void)in_sizes; (void)n_in; (void)out_size;

  // 0. Weight conversions to padded f16 (tiny)
  k_convw<<<(NQKV_ * 64 + 255) / 256, 256, 0, stream>>>(qkv_w, qkvwh, NQKV_, C_, 64, NQKV_ * 64);
  k_convw<<<(C_ * 64 + 255) / 256, 256, 0, stream>>>(proj_w, projwh, C_, C_, 64, C_ * 64);
  k_convw<<<(TPAD * 64 + 255) / 256, 256, 0, stream>>>(w_in, winwh, CH2_, C_, 64, TPAD * 64);
  k_convw<<<(C_ * 128 + 255) / 256, 256, 0, stream>>>(w_out, woutwh, C_, HID_, 128, C_ * 128);

  // 1. LN1
  k_ln1<<<P_ / 256, 256, 0, stream>>>(x, ln1_w, ln1_b, xn1);
  // 2. QKV GEMM: (P/64)*(144/16) = 576*9 waves / 8 per block
  k_gemm_qkv<<<(P_ / 64) * (NQKV_ / 16) / 8, 256, 0, stream>>>(xn1, qkvwh, qkv_b, qkv);
  // 3. Neighborhood attention
  k_attn<<<(P_ * 2) / 256, 256, 0, stream>>>(qkv, rpb, attn);
  // 4/5. ECA gate
  k_eca_pool<<<C_, 256, 0, stream>>>(xn1, pooled);
  k_eca_gate<<<1, 64, 0, stream>>>(pooled, eca_w, gate);
  // 6. Proj GEMM + gate + residual
  k_gemm_proj<<<(P_ / 64) * (C_ / 16) / 8, 256, 0, stream>>>(attn, projwh, proj_b,
                                                             gate, x, xmid);
  // 7. LN2
  k_ln2<<<P_ / 256, 256, 0, stream>>>(xmid, ln2_w, ln2_b, xn2);
  // 8. w_in GEMM -> padded t (no tail guards anywhere)
  k_gemm_win<<<(P_ / 64) * (TPAD / 16) / 8, 256, 0, stream>>>(xn2, winwh, t_h);
  // 9. depthwise + gelu gate
  k_dwgelu<<<(P_ * HID_) / 256, 256, 0, stream>>>(t_h, w_dw, g_h);
  // 10. w_out GEMM + residual -> NCHW out
  k_gemm_wout<<<(P_ / 64) * (C_ / 16) / 8, 256, 0, stream>>>(g_h, w_out == nullptr ? nullptr : woutwh, xmid, out);
}
